// GRU_16381005267367
// MI455X (gfx1250) — matrix operand order repaired and checked by reading, not compiled, not run on hardware
//
#include <hip/hip_runtime.h>
#include <hip/hip_bf16.h>
#include <cstdint>
#include <cstddef>

// Problem dims (fixed by the reference)
#define T_STEPS 512
#define BATCH   64
#define HID     1024
#define NLAYERS 2
#define GATES3  (3 * HID)

typedef __attribute__((ext_vector_type(16))) __bf16 bf16x16;
typedef __attribute__((ext_vector_type(8)))  float  f32x8;

union ABf16 {
    bf16x16 v;
    uint4   q[2];
};

__device__ __forceinline__ unsigned short f32_to_bf16_rn(float f) {
    unsigned int u = __float_as_uint(f);
    u += 0x7FFFu + ((u >> 16) & 1u);   // round to nearest even
    return (unsigned short)(u >> 16);
}

// A-fragment: 16x32 (MxK) bf16, rows of a row-major matrix (stride lda).
// Layout: lane<16 holds row M=lane, K = kBase+0..15;
//         lane>=16 holds row M=lane-16, K = kBase+16..31 (matches B).
__device__ __forceinline__ bf16x16 load_A_frag(const unsigned short* __restrict__ A,
                                               int lda, int tileM, int kBase, int lane) {
    const int half = lane >> 4;
    const int row  = tileM + (lane & 15);
    const unsigned short* p = A + (size_t)row * lda + kBase + half * 16;
    ABf16 a;
    a.q[0] = *(const uint4*)(p);        // K = kBase + 16*half + 0..7
    a.q[1] = *(const uint4*)(p + 8);    // K = kBase + 16*half + 8..15
    return a.v;
}

// B-fragment: 32x16 (KxN) where B[k][n] = W[n][k] with W row-major (stride ldw).
// Layout used here (contiguous k, matches A): lane<16 -> col N=lane, K = kBase+0..15; lane>=16 -> K = kBase+16..31.
// Per lane this is 16 *contiguous* bf16 of W's row n, so "@ W.T" needs no transpose.
__device__ __forceinline__ bf16x16 load_B_frag(const unsigned short* __restrict__ W,
                                               int ldw, int tileN, int kBase, int lane) {
    const int half = lane >> 4;
    const int ncol = tileN + (lane & 15);
    const unsigned short* p = W + (size_t)ncol * ldw + kBase + half * 16;
    ABf16 b;
    b.q[0] = *(const uint4*)(p);
    b.q[1] = *(const uint4*)(p + 8);
    return b.v;
}

__device__ __forceinline__ f32x8 wmma_bf16(bf16x16 a, bf16x16 b, f32x8 c) {
    return __builtin_amdgcn_wmma_f32_16x16x32_bf16(
        /*neg_a=*/false, a, /*neg_b=*/false, b,
        /*c_mod=*/(short)0, c, /*reuse_a=*/false, /*reuse_b=*/false);
}

// ---------------------------------------------------------------------------
// Utility kernels
// ---------------------------------------------------------------------------
__global__ void f32_to_bf16_kernel(const float* __restrict__ in,
                                   unsigned short* __restrict__ out, int n) {
    int i = blockIdx.x * blockDim.x + threadIdx.x;
    if (i < n) out[i] = f32_to_bf16_rn(in[i]);
}

__global__ void zero_h_kernel(float* __restrict__ hf,
                              unsigned short* __restrict__ hbf, int n) {
    int i = blockIdx.x * blockDim.x + threadIdx.x;
    if (i < n) { hf[i] = 0.0f; hbf[i] = 0; }
}

// ---------------------------------------------------------------------------
// gx = hidden @ Wx[l].T + bx[l]   as a (32768 x 3072) GEMM, K = 1024.
// Block = 4 waves covering 64(M) x 64(N); each wave owns a 16x64 strip
// (4 accumulators). Software-pipelined: fragments for k+32 are loaded while
// the 4 WMMAs for k execute, so s_wait_loadcnt retires against loads that
// are already a full iteration old. A rows stream from HBM with WGP-scope
// prefetch-ahead; the B panel (weights) is L2-resident.
// ---------------------------------------------------------------------------
__global__ void __launch_bounds__(128)
gemm_gx_kernel(const unsigned short* __restrict__ A,     // (T*B, H) bf16
               const unsigned short* __restrict__ W,     // (3H, H)  bf16 (Wx[l] flat)
               const float* __restrict__ bias,           // (3H)
               float* __restrict__ gx) {                 // (T*B, 3H)
    const int lane = threadIdx.x & 31;
    const int wave = threadIdx.x >> 5;
    const int blocksN = GATES3 / 64;                      // 48
    const int tileM   = (blockIdx.x / blocksN) * 64 + wave * 16;
    const int tileN0  = (blockIdx.x % blocksN) * 64;

    f32x8 acc[4];
#pragma unroll
    for (int j = 0; j < 4; ++j) {
        const float bval = bias[tileN0 + j * 16 + (lane & 15)];
#pragma unroll
        for (int v = 0; v < 8; ++v) acc[j][v] = bval;
    }

    const unsigned short* aRow = A + (size_t)(tileM + (lane & 15)) * HID;

    // Prologue: fragments for k = 0
    bf16x16 af = load_A_frag(A, HID, tileM, 0, lane);
    bf16x16 b0 = load_B_frag(W, HID, tileN0 +  0, 0, lane);
    bf16x16 b1 = load_B_frag(W, HID, tileN0 + 16, 0, lane);
    bf16x16 b2 = load_B_frag(W, HID, tileN0 + 32, 0, lane);
    bf16x16 b3 = load_B_frag(W, HID, tileN0 + 48, 0, lane);

    for (int k = 0; k < HID - 32; k += 32) {
        const int kn = k + 32;
        __builtin_prefetch(aRow + kn + 224, 0, 3);        // WGP-scope stream-ahead on A
        bf16x16 an = load_A_frag(A, HID, tileM, kn, lane);
        bf16x16 c0 = load_B_frag(W, HID, tileN0 +  0, kn, lane);
        bf16x16 c1 = load_B_frag(W, HID, tileN0 + 16, kn, lane);
        bf16x16 c2 = load_B_frag(W, HID, tileN0 + 32, kn, lane);
        bf16x16 c3 = load_B_frag(W, HID, tileN0 + 48, kn, lane);
        acc[0] = wmma_bf16(af, b0, acc[0]);
        acc[1] = wmma_bf16(af, b1, acc[1]);
        acc[2] = wmma_bf16(af, b2, acc[2]);
        acc[3] = wmma_bf16(af, b3, acc[3]);
        af = an; b0 = c0; b1 = c1; b2 = c2; b3 = c3;
    }
    // Epilogue: last K-slice
    acc[0] = wmma_bf16(af, b0, acc[0]);
    acc[1] = wmma_bf16(af, b1, acc[1]);
    acc[2] = wmma_bf16(af, b2, acc[2]);
    acc[3] = wmma_bf16(af, b3, acc[3]);

    const int rowBase = tileM + ((lane >> 4) << 3);
#pragma unroll
    for (int j = 0; j < 4; ++j) {
        const int col = tileN0 + j * 16 + (lane & 15);
#pragma unroll
        for (int v = 0; v < 8; ++v)
            gx[(size_t)(rowBase + v) * GATES3 + col] = acc[j][v];
    }
}

// ---------------------------------------------------------------------------
// Step 1 (per timestep): r = sigmoid(gr + h@Whr.T), z = sigmoid(gz + h@Whz.T)
// emits z (f32) and h*r (bf16). One wave per 16(M) x 32(N) strip, both gates.
// Software-pipelined like the big GEMM (this is the serial latency chain).
// ---------------------------------------------------------------------------
__global__ void __launch_bounds__(128)
gru_step1_kernel(const float* __restrict__ gx_t,          // (B, 3H)
                 const unsigned short* __restrict__ hbf,  // (B, H) bf16
                 const float* __restrict__ hf,            // (B, H) f32
                 const unsigned short* __restrict__ Whr,  // (H, H) bf16
                 const unsigned short* __restrict__ Whz,  // (H, H) bf16
                 float* __restrict__ zbuf,                // (B, H)
                 unsigned short* __restrict__ hrbf) {     // (B, H) bf16
    const int lane  = threadIdx.x & 31;
    const int wid   = blockIdx.x * 4 + (threadIdx.x >> 5);   // 0..127
    const int tileM  = (wid >> 5) * 16;                       // 4 tiles over B
    const int tileN0 = (wid & 31) * 32;                       // 32 strips over H
    const int rowBase = tileM + ((lane >> 4) << 3);

    f32x8 accR[2], accZ[2];
#pragma unroll
    for (int j = 0; j < 2; ++j) {
        const int col = tileN0 + j * 16 + (lane & 15);
#pragma unroll
        for (int v = 0; v < 8; ++v) {
            accR[j][v] = gx_t[(size_t)(rowBase + v) * GATES3 + 0 * HID + col];
            accZ[j][v] = gx_t[(size_t)(rowBase + v) * GATES3 + 1 * HID + col];
        }
    }

    bf16x16 af  = load_A_frag(hbf, HID, tileM, 0, lane);
    bf16x16 br0 = load_B_frag(Whr, HID, tileN0 +  0, 0, lane);
    bf16x16 br1 = load_B_frag(Whr, HID, tileN0 + 16, 0, lane);
    bf16x16 bz0 = load_B_frag(Whz, HID, tileN0 +  0, 0, lane);
    bf16x16 bz1 = load_B_frag(Whz, HID, tileN0 + 16, 0, lane);

    for (int k = 0; k < HID - 32; k += 32) {
        const int kn = k + 32;
        bf16x16 an = load_A_frag(hbf, HID, tileM, kn, lane);
        bf16x16 cr0 = load_B_frag(Whr, HID, tileN0 +  0, kn, lane);
        bf16x16 cr1 = load_B_frag(Whr, HID, tileN0 + 16, kn, lane);
        bf16x16 cz0 = load_B_frag(Whz, HID, tileN0 +  0, kn, lane);
        bf16x16 cz1 = load_B_frag(Whz, HID, tileN0 + 16, kn, lane);
        accR[0] = wmma_bf16(af, br0, accR[0]);
        accR[1] = wmma_bf16(af, br1, accR[1]);
        accZ[0] = wmma_bf16(af, bz0, accZ[0]);
        accZ[1] = wmma_bf16(af, bz1, accZ[1]);
        af = an; br0 = cr0; br1 = cr1; bz0 = cz0; bz1 = cz1;
    }
    accR[0] = wmma_bf16(af, br0, accR[0]);
    accR[1] = wmma_bf16(af, br1, accR[1]);
    accZ[0] = wmma_bf16(af, bz0, accZ[0]);
    accZ[1] = wmma_bf16(af, bz1, accZ[1]);

#pragma unroll
    for (int j = 0; j < 2; ++j) {
        const int col = tileN0 + j * 16 + (lane & 15);
#pragma unroll
        for (int v = 0; v < 8; ++v) {
            const int idx = (rowBase + v) * HID + col;
            const float r = 1.0f / (1.0f + __expf(-accR[j][v]));
            const float z = 1.0f / (1.0f + __expf(-accZ[j][v]));
            zbuf[idx] = z;
            hrbf[idx] = f32_to_bf16_rn(hf[idx] * r);
        }
    }
}

// ---------------------------------------------------------------------------
// Step 2 (per timestep): c = tanh(gc + (h*r)@Whh.T); h' = z*h + (1-z)*c
// One wave per 16x32 strip, software-pipelined.
// ---------------------------------------------------------------------------
__global__ void __launch_bounds__(128)
gru_step2_kernel(const float* __restrict__ gx_t,           // (B, 3H)
                 const unsigned short* __restrict__ hrbf,  // (B, H) bf16
                 const unsigned short* __restrict__ Whh,   // (H, H) bf16
                 const float* __restrict__ zbuf,           // (B, H)
                 float* __restrict__ hf,                   // (B, H) f32 in/out
                 unsigned short* __restrict__ hbf,         // (B, H) bf16 out
                 float* __restrict__ y_t) {                // (B, H) layer output at t
    const int lane  = threadIdx.x & 31;
    const int wid   = blockIdx.x * 4 + (threadIdx.x >> 5);
    const int tileM  = (wid >> 5) * 16;
    const int tileN0 = (wid & 31) * 32;
    const int rowBase = tileM + ((lane >> 4) << 3);

    f32x8 accC[2];
#pragma unroll
    for (int j = 0; j < 2; ++j) {
        const int col = tileN0 + j * 16 + (lane & 15);
#pragma unroll
        for (int v = 0; v < 8; ++v)
            accC[j][v] = gx_t[(size_t)(rowBase + v) * GATES3 + 2 * HID + col];
    }

    bf16x16 af = load_A_frag(hrbf, HID, tileM, 0, lane);
    bf16x16 b0 = load_B_frag(Whh, HID, tileN0 +  0, 0, lane);
    bf16x16 b1 = load_B_frag(Whh, HID, tileN0 + 16, 0, lane);

    for (int k = 0; k < HID - 32; k += 32) {
        const int kn = k + 32;
        bf16x16 an = load_A_frag(hrbf, HID, tileM, kn, lane);
        bf16x16 c0 = load_B_frag(Whh, HID, tileN0 +  0, kn, lane);
        bf16x16 c1 = load_B_frag(Whh, HID, tileN0 + 16, kn, lane);
        accC[0] = wmma_bf16(af, b0, accC[0]);
        accC[1] = wmma_bf16(af, b1, accC[1]);
        af = an; b0 = c0; b1 = c1;
    }
    accC[0] = wmma_bf16(af, b0, accC[0]);
    accC[1] = wmma_bf16(af, b1, accC[1]);

#pragma unroll
    for (int j = 0; j < 2; ++j) {
        const int col = tileN0 + j * 16 + (lane & 15);
#pragma unroll
        for (int v = 0; v < 8; ++v) {
            const int idx = (rowBase + v) * HID + col;
            const float c  = tanhf(accC[j][v]);
            const float z  = zbuf[idx];
            const float h0 = hf[idx];
            const float hn = z * h0 + (1.0f - z) * c;
            hf[idx]  = hn;
            hbf[idx] = f32_to_bf16_rn(hn);
            y_t[idx] = hn;
        }
    }
}

// ---------------------------------------------------------------------------
// Host side
// ---------------------------------------------------------------------------
extern "C" void kernel_launch(void* const* d_in, const int* in_sizes, int n_in,
                              void* d_out, int out_size, void* d_ws, size_t ws_size,
                              hipStream_t stream) {
    (void)in_sizes; (void)n_in; (void)out_size; (void)ws_size;

    const float* x  = (const float*)d_in[0];   // (T, B, H)
    const float* Wx = (const float*)d_in[1];   // (L, 3, H, H)
    const float* bx = (const float*)d_in[2];   // (L, 3, H)
    const float* Wh = (const float*)d_in[3];   // (L, 3, H, H)

    float* out_hidden = (float*)d_out;                                     // (T,B,H)
    float* out_states = out_hidden + (size_t)T_STEPS * BATCH * HID;        // (L,B,H)

    // Workspace carve-out (256B aligned)
    char* ws = (char*)d_ws;
    size_t off = 0;
    auto carve = [&](size_t bytes) -> void* {
        void* p = ws + off;
        off += (bytes + 255) & ~(size_t)255;
        return p;
    };
    const size_t nWx  = (size_t)NLAYERS * 3 * HID * HID;      // 6.29M elems
    const size_t nSeq = (size_t)T_STEPS * BATCH * HID;        // 33.55M elems
    unsigned short* Wx_bf   = (unsigned short*)carve(nWx * 2);
    unsigned short* Wh_bf   = (unsigned short*)carve(nWx * 2);
    unsigned short* hA_bf   = (unsigned short*)carve(nSeq * 2);            // GEMM A operand
    float*          gx      = (float*)carve(nSeq * 3 * 4);                 // (T*B, 3H)
    float*          hidden0 = (float*)carve(nSeq * 4);                     // layer-0 output seq
    float*          hf      = (float*)carve((size_t)BATCH * HID * 4);
    unsigned short* hbf     = (unsigned short*)carve((size_t)BATCH * HID * 2);
    unsigned short* hrbf    = (unsigned short*)carve((size_t)BATCH * HID * 2);
    float*          zbuf    = (float*)carve((size_t)BATCH * HID * 4);

    // One-time weight conversion to bf16
    {
        int n = (int)nWx;
        f32_to_bf16_kernel<<<(n + 255) / 256, 256, 0, stream>>>(Wx, Wx_bf, n);
        f32_to_bf16_kernel<<<(n + 255) / 256, 256, 0, stream>>>(Wh, Wh_bf, n);
    }

    const int gemmGrid = ((T_STEPS * BATCH) / 64) * (GATES3 / 64);   // 512*48 = 24576
    const int stepGrid = ((BATCH / 16) * (HID / 32)) / 4;            // 128 waves / 4 = 32
    const int nBH      = BATCH * HID;

    for (int l = 0; l < NLAYERS; ++l) {
        const float* hidden_in = (l == 0) ? x : hidden0;
        float*       ys        = (l == NLAYERS - 1) ? out_hidden : hidden0;

        // Layer input -> bf16 A operand, then batched gx GEMM (+bias)
        f32_to_bf16_kernel<<<((int)nSeq + 255) / 256, 256, 0, stream>>>(
            hidden_in, hA_bf, (int)nSeq);
        gemm_gx_kernel<<<gemmGrid, 128, 0, stream>>>(
            hA_bf, Wx_bf + (size_t)l * 3 * HID * HID, bx + (size_t)l * GATES3, gx);

        // h = 0
        zero_h_kernel<<<(nBH + 255) / 256, 256, 0, stream>>>(hf, hbf, nBH);

        const unsigned short* Whl = Wh_bf + (size_t)l * 3 * HID * HID;
        const unsigned short* Whr = Whl;
        const unsigned short* Whz = Whl + (size_t)HID * HID;
        const unsigned short* Whh = Whl + (size_t)2 * HID * HID;

        for (int t = 0; t < T_STEPS; ++t) {
            const float* gx_t = gx + (size_t)t * BATCH * GATES3;
            gru_step1_kernel<<<stepGrid, 128, 0, stream>>>(
                gx_t, hbf, hf, Whr, Whz, zbuf, hrbf);
            gru_step2_kernel<<<stepGrid, 128, 0, stream>>>(
                gx_t, hrbf, Whh, zbuf, hf, hbf, ys + (size_t)t * BATCH * HID);
        }

        // states[l] = h_last
        hipMemcpyAsync(out_states + (size_t)l * BATCH * HID, hf,
                       (size_t)BATCH * HID * sizeof(float),
                       hipMemcpyDeviceToDevice, stream);
    }
}
